// pRecurrentLayer_1778116461269
// MI455X (gfx1250) — matrix operand order repaired and checked by reading, not compiled, not run on hardware
//
#include <hip/hip_runtime.h>
#include <hip/hip_bf16.h>

// ---------------------------------------------------------------------------
// pRecurrentLayer for MI455X (gfx1250, wave32).
//   K1 setup:   fold pmac noise into effective weights; pack W2 into the
//               native v_wmma_f32_16x16x32_f16 B-fragment layout (f16).
//   K2 pmac1:   small K=32 GEMM + tanh activation (VALU), writes a1 (f32)
//               and feature column j=0 of h (f16).
//   K3 filter:  fused double RC filter-bank recurrence over T=256, one lane
//               per (c,f,n,b); writes features j=1..32 of h (f16).
//   K4 pmac2:   M=8192/n, K=1056, N=32 GEMM via v_wmma_f32_16x16x32_f16,
//               f32 accum, fused bias + tanh activation, writes d_out.
// h (69 MB f16) is L2-resident on MI455X (192 MB L2), so the big
// intermediate never needs to round-trip HBM.
// ---------------------------------------------------------------------------

typedef _Float16 v16h __attribute__((ext_vector_type(16)));
typedef float    v8f  __attribute__((ext_vector_type(8)));

#define N_VAR 4
#define BATCH 32
#define N_IN  32
#define N_OUT 32
#define N_FEAT 32
#define T_LEN 256
#define KFEAT 1056            // N_OUT * (N_FEAT + 1)
#define EPSN  0.05f
#define COUPLING 0.2f
#define DT_C  0.1f
#define RMIN_C 1.0e5f
#define RMAX_C 1.0e7f
#define CMIN_C 1.0e-7f
#define CMAX_C 1.0e-4f

// salts for independent hash-RNG streams
#define S_TH1 0x11111111u
#define S_TH2 0x22222222u
#define S_R1  0x33333331u
#define S_C1  0x33333332u
#define S_M1  0x33333333u
#define S_S1  0x33333334u
#define S_R2  0x44444441u
#define S_C2  0x44444442u
#define S_M2  0x44444443u
#define S_S2  0x44444444u

__device__ __forceinline__ float u01(uint32_t x) {
  x ^= x >> 17; x *= 0xed5ad4bbu;
  x ^= x >> 11; x *= 0xac4c1b51u;
  x ^= x >> 15; x *= 0x31848babu;
  x ^= x >> 14;
  return (float)(x >> 8) * (1.0f / 16777216.0f);
}

__device__ __forceinline__ float theta_eff(float t) {
  t = fminf(fmaxf(t, -10.0f), 10.0f);
  return (fabsf(t) < 0.01f) ? 0.0f : t;
}

// thn = theta_eff * multiplicative noise, per-variant n
__device__ __forceinline__ float thn_val(const float* th, int rows, int n,
                                         int m, int o, uint32_t salt) {
  float t  = theta_eff(th[m * N_OUT + o]);
  float nz = (u01(salt + (uint32_t)((n * rows + m) * N_OUT + o)) * 2.0f - 1.0f)
             * EPSN + 1.0f;
  return t * nz;
}

__device__ __forceinline__ float actf(float z) {
  return 0.05f + 0.5f * tanhf((z - 0.3f) * 3.0f);
}

__device__ __forceinline__ float sigmoidf_(float x) {
  return 1.0f / (1.0f + __expf(-x));
}

// ---------------------------------------------------------------------------
// K1: build effective weights.  block 0: theta1 -> W1eff[n][33][32] f32.
//     block 1: theta2 -> Bpack (WMMA B-fragment f16 layout) + bias2 f32.
// Bpack layout: [n][kk(33)][ot(2)][lane(32)][v(8)] dwords; lane=(g,col),
// dword v packs K rows (8g + 2v, +1) for v<4, (16 + 8g + 2(v-4), +1) for v>=4, of column o = ot*16+col.
// ---------------------------------------------------------------------------
__global__ void setup_weights(const float* __restrict__ th1,
                              const float* __restrict__ th2,
                              float* __restrict__ W1,
                              uint32_t* __restrict__ Bpack,
                              float* __restrict__ bias2) {
  const int tid = threadIdx.x;           // 128 threads: (n,o)
  const int n = tid >> 5, o = tid & 31;
  if (blockIdx.x == 0) {
    float denom = 1e-10f;
    for (int m = 0; m < N_IN + 2; ++m)
      denom += fabsf(thn_val(th1, N_IN + 2, n, m, o, S_TH1));
    const float inv = 1.0f / denom;
    for (int m = 0; m < N_IN + 1; ++m)   // rows 0..31 weights, row 32 bias
      W1[(n * 33 + m) * N_OUT + o] = thn_val(th1, N_IN + 2, n, m, o, S_TH1) * inv;
  } else {
    const int rows = KFEAT + 2;          // 1058
    float denom = 1e-10f;
    for (int m = 0; m < rows; ++m)
      denom += fabsf(thn_val(th2, rows, n, m, o, S_TH2));
    const float inv = 1.0f / denom;
    const int col = o & 15, ot = o >> 4;
    for (int kk = 0; kk < 33; ++kk) {
      for (int krel = 0; krel < 32; krel += 2) {
        const int m = kk * 32 + krel;
        union { _Float16 h[2]; uint32_t u; } pk;
        pk.h[0] = (_Float16)(thn_val(th2, rows, n, m,     o, S_TH2) * inv);
        pk.h[1] = (_Float16)(thn_val(th2, rows, n, m + 1, o, S_TH2) * inv);
        const int g = (krel >> 3) & 1, v = ((krel & 7) >> 1) + ((krel >> 4) << 2);
        Bpack[(((n * 33 + kk) * 2 + ot) * 32 + (g * 16 + col)) * 8 + v] = pk.u;
      }
    }
    bias2[n * N_OUT + o] = thn_val(th2, rows, n, KFEAT, o, S_TH2) * inv;
  }
}

// ---------------------------------------------------------------------------
// K2: pmac1 (K=32) -> a1[n][b][c][t] f32 and h feature j=0 (f16).
// ---------------------------------------------------------------------------
__global__ void pmac1_kernel(const float* __restrict__ x,
                             const float* __restrict__ W1,
                             float* __restrict__ a1,
                             _Float16* __restrict__ h) {
  const int idx = blockIdx.x * blockDim.x + threadIdx.x;   // 1,048,576
  const int t = idx & 255;
  const int c = (idx >> 8) & 31;
  const int b = (idx >> 13) & 31;
  const int n = idx >> 18;
  const float* xp = x + ((size_t)(n * 32 + b) * 32) * 256 + t;
  const float* wp = W1 + (size_t)(n * 33) * 32 + c;
  float z = wp[32 * 32];                 // bias row (m = 32)
#pragma unroll
  for (int m = 0; m < 32; ++m) z = fmaf(xp[m * 256], wp[m * 32], z);
  const float a = actf(z);
  a1[((size_t)((n * 32 + b) * 32 + c)) * 256 + t] = a;
  h[((size_t)((n * 32 + b) * 256 + t)) * KFEAT + c * 33] = (_Float16)a;
}

// ---------------------------------------------------------------------------
// K3: fused double filter bank.  One lane per (c, f, n, b); 256-step scan.
// Scan output is the PRE-update state (y[t] = s before absorbing x[t]).
// ---------------------------------------------------------------------------
__device__ __forceinline__ float beta_of(uint32_t base, uint32_t sR, uint32_t sC,
                                         uint32_t sM, float Rt, float Ct) {
  const float nR = (u01(sR + base) * 2.0f - 1.0f) * EPSN + 1.0f;
  const float nC = (u01(sC + base) * 2.0f - 1.0f) * EPSN + 1.0f;
  const float mu = u01(sM + base) * COUPLING + 1.0f;
  const float RC = mu * (Rt * nR) * (Ct * nC);
  return RC / (RC + DT_C);
}

__global__ void filter_kernel(const float* __restrict__ a1,
                              const float* __restrict__ Rlf,
                              const float* __restrict__ Clf,
                              _Float16* __restrict__ h) {
  const int w = blockIdx.x * 8 + (threadIdx.x >> 5);   // 4096 waves
  const int f = threadIdx.x & 31;
  const int c = w >> 7, nn = (w >> 5) & 3, b = w & 31;
  const int k = c * 32 + f;

  const float Rt0 = sigmoidf_(Rlf[k * 2 + 0]) * (RMAX_C - RMIN_C) + RMIN_C;
  const float Rt1 = sigmoidf_(Rlf[k * 2 + 1]) * (RMAX_C - RMIN_C) + RMIN_C;
  const float Ct0 = sigmoidf_(Clf[k * 2 + 0]) * (CMAX_C - CMIN_C) + CMIN_C;
  const float Ct1 = sigmoidf_(Clf[k * 2 + 1]) * (CMAX_C - CMIN_C) + CMIN_C;

  float s1 = u01(S_S1 + (uint32_t)((k * 4 + nn) * 32 + b));
  float s2 = u01(S_S2 + (uint32_t)((k * 4 + nn) * 32 + b));

  const float* xp = a1 + ((size_t)((nn * 32 + b) * 32 + c)) * 256;
  _Float16* hp = h + (size_t)((nn * 32 + b) * 256) * KFEAT + c * 33 + 1 + f;

  for (int t = 0; t < 256; ++t) {
    const uint32_t base = ((uint32_t)t * 1024u + (uint32_t)k) * 4u + (uint32_t)nn;
    const float b1 = beta_of(base, S_R1, S_C1, S_M1, Rt0, Ct0);
    const float b2 = beta_of(base, S_R2, S_C2, S_M2, Rt1, Ct1);
    const float y1 = s1;                 // bank-1 output (pre-update state)
    const float y2 = s2;                 // bank-2 output (pre-update state)
    s2 = b2 * s2 + (1.0f - b2) * y1;     // bank 2 absorbs delayed bank-1 state
    s1 = b1 * s1 + (1.0f - b1) * xp[t];  // bank 1 absorbs input
    hp[(size_t)t * KFEAT] = (_Float16)y2;
  }
}

// ---------------------------------------------------------------------------
// K4: pmac2 via WMMA.  Per wave: 16 rows x 32 cols, K loop of 33 x 32.
// A fragment: two contiguous 16B chunks of an h row (16-bit A 16x32 layout).
// B fragment: prepacked, one contiguous 32B load per lane.
// ---------------------------------------------------------------------------
__global__ void pmac2_wmma(const _Float16* __restrict__ h,
                           const uint32_t* __restrict__ Bpack,
                           const float* __restrict__ bias2,
                           float* __restrict__ out) {
  const int lane = threadIdx.x & 31;
  const int wv = threadIdx.x >> 5;
  const int n = blockIdx.x >> 6;
  const int rowTile = (blockIdx.x & 63) * 8 + wv;   // 0..511
  const int r0 = rowTile * 16;
  const int g = lane >> 4;                           // lane group
  const int mrow = lane & 15;
  const int r = r0 + mrow;                           // all 16 rows share b
  const int b = r >> 8, t = r & 255;

  const _Float16* arow = h + ((size_t)((n * 32 + b) * 256 + t)) * KFEAT;
  const uint32_t* bbase = Bpack + ((size_t)(n * 33) * 2 * 32 + (size_t)lane) * 8;

  v8f c0 = {0.f,0.f,0.f,0.f,0.f,0.f,0.f,0.f};
  v8f c1 = {0.f,0.f,0.f,0.f,0.f,0.f,0.f,0.f};

  for (int kk = 0; kk < 33; ++kk) {
    const char* ab = (const char*)arow + (kk * 64 + g * 16);
    v16h a;
    ((uint4*)&a)[0] = *(const uint4*)(ab);
    ((uint4*)&a)[1] = *(const uint4*)(ab + 32);

    const uint32_t* bp = bbase + (size_t)kk * 512;   // [kk][ot][lane][v]
    v16h b0v, b1v;
    ((uint4*)&b0v)[0] = *(const uint4*)(bp);
    ((uint4*)&b0v)[1] = *(const uint4*)(bp + 4);
    ((uint4*)&b1v)[0] = *(const uint4*)(bp + 256);
    ((uint4*)&b1v)[1] = *(const uint4*)(bp + 260);

    if (kk + 1 < 33)                                 // -> global_prefetch_b8
      __builtin_prefetch((const char*)arow + ((kk + 1) * 64 + g * 16), 0, 1);

    c0 = __builtin_amdgcn_wmma_f32_16x16x32_f16(false, a, false, b0v,
                                                (short)0, c0, false, false);
    c1 = __builtin_amdgcn_wmma_f32_16x16x32_f16(false, a, false, b1v,
                                                (short)0, c1, false, false);
  }

  const int o0 = lane & 15;
  const float bz0 = bias2[n * 32 + o0];
  const float bz1 = bias2[n * 32 + 16 + o0];
#pragma unroll
  for (int i = 0; i < 8; ++i) {
    const int rr = r0 + g * 8 + i;                   // D: M = 8*g + i
    const int bb = rr >> 8, tt = rr & 255;
    out[((size_t)((n * 32 + bb) * 32 + o0)) * 256 + tt]      = actf(c0[i] + bz0);
    out[((size_t)((n * 32 + bb) * 32 + 16 + o0)) * 256 + tt] = actf(c1[i] + bz1);
  }
}

// ---------------------------------------------------------------------------
extern "C" void kernel_launch(void* const* d_in, const int* in_sizes, int n_in,
                              void* d_out, int out_size, void* d_ws, size_t ws_size,
                              hipStream_t stream) {
  const float* x    = (const float*)d_in[0];   // (4,32,32,256)
  const float* th1  = (const float*)d_in[1];   // (34,32)
  const float* th2  = (const float*)d_in[2];   // (1058,32)
  const float* Rlf  = (const float*)d_in[3];   // (32,32,2)
  const float* Clf  = (const float*)d_in[4];   // (32,32,2)
  float* out = (float*)d_out;                  // (4,32,32,256)

  char* ws = (char*)d_ws;
  float*    W1    = (float*)(ws);                  // 4*33*32 f32   (16.9 KB)
  float*    bias2 = (float*)(ws + 0x8000);         // 128 f32
  uint32_t* Bpack = (uint32_t*)(ws + 0x10000);     // 67584 u32     (270 KB)
  float*    a1    = (float*)(ws + 0x60000);        // 1,048,576 f32 (4 MB)
  _Float16* hbuf  = (_Float16*)(ws + 0x500000);    // 34,603,008 f16 (69 MB)

  setup_weights<<<2, 128, 0, stream>>>(th1, th2, W1, Bpack, bias2);
  pmac1_kernel<<<4096, 256, 0, stream>>>(x, W1, a1, hbuf);
  filter_kernel<<<512, 256, 0, stream>>>(a1, Rlf, Clf, hbuf);
  pmac2_wmma<<<256, 256, 0, stream>>>(hbuf, Bpack, bias2, out);
}
